// ResContextBlock_5016521801757
// MI455X (gfx1250) — matrix operand order repaired and checked by reading, not compiled, not run on hardware
//
#include <hip/hip_runtime.h>
#include <hip/hip_bf16.h>
#include <stdint.h>

// ---------------------------------------------------------------------------
// CDNA5 (gfx1250) wave32 WMMA sparse-conv ResContextBlock.
// Matrix ops in bf16 (v_wmma_f32_16x16x32_bf16), fp32 accumulate + BN stats.
// Weight B-fragments staged in WGP LDS (ds_load_b128 in the hot loop).
// ---------------------------------------------------------------------------

typedef __attribute__((ext_vector_type(16))) __bf16 v16bf;
typedef __attribute__((ext_vector_type(8)))  float  v8f;

#define NN   200000
#define EPSV 1e-5f

struct U8 { uint4 a, b; };

__device__ __forceinline__ uint16_t f2bf(float f) {
  uint32_t u = __builtin_bit_cast(uint32_t, f);
  u += 0x7FFFu + ((u >> 16) & 1u);   // round-to-nearest-even
  return (uint16_t)(u >> 16);
}

// Zero the BN statistic accumulators (4 convs x 256 floats).
__global__ void k_init(float* stats) {
  int i = blockIdx.x * blockDim.x + threadIdx.x;
  if (i < 1024) stats[i] = 0.f;
}

// fp32 feats [N,32] -> bf16 [(N+1),32] with zero sentinel row N.
__global__ void k_pack_x(const float* __restrict__ src, uint16_t* __restrict__ dst,
                         int n, int c) {
  int i = blockIdx.x * blockDim.x + threadIdx.x;
  int total = (n + 1) * c;
  if (i >= total) return;
  int row = i / c;
  dst[i] = (row < n) ? f2bf(src[i]) : (uint16_t)0;
}

// Pack W[9,Cin,64] fp32 into the wave32 B-fragment layout:
// dst[chunk][tile][lane][half] bf16; half<8 -> K=8*(lane/16)+half, half>=8 -> K=16+8*(lane/16)+half-8.
__global__ void k_pack_w(const float* __restrict__ w, uint16_t* __restrict__ dst, int cin) {
  int chunks = (cin == 32) ? 9 : 18;
  int i = blockIdx.x * blockDim.x + threadIdx.x;
  if (i >= chunks * 2048) return;
  int h    = i & 15;
  int lane = (i >> 4) & 31;
  int t    = (i >> 9) & 3;
  int j    = i >> 11;
  int K    = (h < 8) ? (((lane >> 4) << 3) + h) : (16 + ((lane >> 4) << 3) + (h - 8));
  int gk   = j * 32 + K;                // global K index in concat(offset, channel)
  int koff = gk / cin;
  int c    = gk - koff * cin;
  int col  = (t << 4) + (lane & 15);
  dst[i] = f2bf(w[(koff * cin + c) * 64 + col]);
}

// Gather-GEMM: one wave per 16-row output tile, Cout=64 via 4 WMMA col-tiles.
// B fragments staged once per workgroup in LDS; A gathered per-tile from global.
// Applies LeakyReLU and accumulates per-channel sum / sumsq for BatchNorm.
template <int CIN>
__global__ void __launch_bounds__(256)
k_conv(const uint16_t* __restrict__ x,      // bf16 [(N+1)*CIN]
       const long long* __restrict__ nbr,   // [N*9], sentinel = N
       const uint16_t* __restrict__ wpk,    // packed bf16 B frags [CHUNKS*4*32*16]
       float* __restrict__ y,               // fp32 [N*64]
       float* __restrict__ stats) {         // [0:64]=sum [64:128]=sumsq
  constexpr int CHUNKS = 9 * CIN / 32;
  __shared__ uint4 wlds[CHUNKS * 256];      // CHUNKS*4 frags * 32 lanes * 32B

  // Cooperative stage: global -> LDS (all 256 threads, before any divergence).
  {
    const uint4* gw = reinterpret_cast<const uint4*>(wpk);
    for (int i = threadIdx.x; i < CHUNKS * 256; i += 256) wlds[i] = gw[i];
  }
  __syncthreads();

  const int wave = threadIdx.x >> 5;
  const int lane = threadIdx.x & 31;
  const int tile = blockIdx.x * (blockDim.x >> 5) + wave;
  if (tile >= (NN >> 4)) return;                  // wave-uniform: EXEC stays all-1s
  const int r    = lane & 15;
  const int row  = (tile << 4) + r;
  const int hi16 = lane >> 4;                     // 0 | 1
  const int kb   = hi16 << 3;                     // A-frag K base: 0 | 8
  const v16bf* wl = reinterpret_cast<const v16bf*>(wlds);

  v8f acc0 = {}, acc1 = {}, acc2 = {}, acc3 = {};
#pragma unroll
  for (int j = 0; j < CHUNKS; ++j) {
    const int koff = (CIN == 32) ? j : (j >> 1);
    const int cb   = (CIN == 32) ? 0 : ((j & 1) << 5);
    const long long idx = nbr[row * 9 + koff];    // sentinel row N is all-zero
    const uint16_t* xr = x + idx * CIN + cb + kb;
    U8 u;
    u.a = *reinterpret_cast<const uint4*>(xr);       // K = kb .. kb+7
    u.b = *reinterpret_cast<const uint4*>(xr + 16);  // K = kb+16 .. kb+23
    const v16bf a  = __builtin_bit_cast(v16bf, u);
    const v16bf b0 = wl[(j * 4 + 0) * 32 + lane];
    const v16bf b1 = wl[(j * 4 + 1) * 32 + lane];
    const v16bf b2 = wl[(j * 4 + 2) * 32 + lane];
    const v16bf b3 = wl[(j * 4 + 3) * 32 + lane];
    acc0 = __builtin_amdgcn_wmma_f32_16x16x32_bf16(false, a, false, b0, (short)0, acc0, false, false);
    acc1 = __builtin_amdgcn_wmma_f32_16x16x32_bf16(false, a, false, b1, (short)0, acc1, false, false);
    acc2 = __builtin_amdgcn_wmma_f32_16x16x32_bf16(false, a, false, b2, (short)0, acc2, false, false);
    acc3 = __builtin_amdgcn_wmma_f32_16x16x32_bf16(false, a, false, b3, (short)0, acc3, false, false);
  }

  // Epilogue: D layout = VGPR v -> row (v + 8*hi16), col = t*16 + (lane&15).
  const int mBase = hi16 << 3;
  v8f accs[4] = {acc0, acc1, acc2, acc3};
#pragma unroll
  for (int t = 0; t < 4; ++t) {
    const int col = (t << 4) + r;
    float s = 0.f, sq = 0.f;
#pragma unroll
    for (int v = 0; v < 8; ++v) {
      float val = accs[t][v];
      val = (val > 0.f) ? val : 0.01f * val;      // LeakyReLU(0.01)
      y[((tile << 4) + mBase + v) * 64 + col] = val;
      s += val;  sq += val * val;
    }
    atomicAdd(&stats[col], s);
    atomicAdd(&stats[64 + col], sq);
  }
}

// Per-channel BN: scale = gamma*rsqrt(var+eps), shift = beta - mean*scale.
__global__ void k_bn_fin(float* stats, const float* __restrict__ g,
                         const float* __restrict__ b) {
  int c = threadIdx.x;
  if (c >= 64) return;
  const float inv_n = 1.f / (float)NN;
  float mean = stats[c] * inv_n;
  float var  = stats[64 + c] * inv_n - mean * mean;   // biased variance
  float sc   = g[c] * rsqrtf(var + EPSV);
  stats[128 + c] = sc;
  stats[192 + c] = b[c] - mean * sc;
}

// Normalize fp32 activations -> bf16 feed for next conv (+ zero sentinel row).
__global__ void k_bn_apply(const float* __restrict__ y, const float* __restrict__ stats,
                           uint16_t* __restrict__ act) {
  int i = blockIdx.x * blockDim.x + threadIdx.x;
  if (i >= (NN + 1) * 64) return;
  int c   = i & 63;
  int row = i >> 6;
  if (row < NN) {
    act[i] = f2bf(y[i] * stats[128 + c] + stats[192 + c]);
  } else {
    act[i] = 0;
  }
}

// out = BN(shortcut conv2) + BN(resA conv4), fused in fp32.
__global__ void k_final(const float* __restrict__ ySc, const float* __restrict__ stSc,
                        const float* __restrict__ yRa, const float* __restrict__ stRa,
                        float* __restrict__ out) {
  int i = blockIdx.x * blockDim.x + threadIdx.x;
  if (i >= NN * 64) return;
  int c = i & 63;
  out[i] = (ySc[i] * stSc[128 + c] + stSc[192 + c]) +
           (yRa[i] * stRa[128 + c] + stRa[192 + c]);
}

extern "C" void kernel_launch(void* const* d_in, const int* in_sizes, int n_in,
                              void* d_out, int out_size, void* d_ws, size_t ws_size,
                              hipStream_t stream) {
  (void)in_sizes; (void)n_in; (void)out_size; (void)ws_size;
  const float*     feats  = (const float*)d_in[0];
  const long long* nbr133 = (const long long*)d_in[1];
  const long long* nbr313 = (const long long*)d_in[2];
  const float* W1 = (const float*)d_in[3];
  const float* W2 = (const float*)d_in[4];
  const float* W3 = (const float*)d_in[5];
  const float* W4 = (const float*)d_in[6];
  const float* g0  = (const float*)d_in[7];
  const float* b0  = (const float*)d_in[8];
  const float* g02 = (const float*)d_in[9];
  const float* b02 = (const float*)d_in[10];
  const float* g1  = (const float*)d_in[11];
  const float* b1  = (const float*)d_in[12];
  const float* g2  = (const float*)d_in[13];
  const float* b2  = (const float*)d_in[14];
  float* out = (float*)d_out;

  char* ws = (char*)d_ws;
  size_t o = 0;
  auto alloc = [&](size_t bytes) -> void* {
    void* p = ws + o;
    o += bytes;
    o = (o + 255) & ~(size_t)255;
    return p;
  };
  uint16_t* xbf = (uint16_t*)alloc((size_t)(NN + 1) * 32 * 2);
  uint16_t* act = (uint16_t*)alloc((size_t)(NN + 1) * 64 * 2);
  float*    yA  = (float*)   alloc((size_t)NN * 64 * 4);
  float*    yB  = (float*)   alloc((size_t)NN * 64 * 4);
  uint16_t* wp1 = (uint16_t*)alloc((size_t) 9 * 2048 * 2);
  uint16_t* wp2 = (uint16_t*)alloc((size_t)18 * 2048 * 2);
  uint16_t* wp3 = (uint16_t*)alloc((size_t) 9 * 2048 * 2);
  uint16_t* wp4 = (uint16_t*)alloc((size_t)18 * 2048 * 2);
  float*  stats = (float*)   alloc(1024 * 4);
  float* st1 = stats;       float* st2 = stats + 256;
  float* st3 = stats + 512; float* st4 = stats + 768;

  k_init<<<4, 256, 0, stream>>>(stats);
  { int tot = (NN + 1) * 32;
    k_pack_x<<<(tot + 255) / 256, 256, 0, stream>>>(feats, xbf, NN, 32); }
  k_pack_w<<<( 9 * 2048 + 255) / 256, 256, 0, stream>>>(W1, wp1, 32);
  k_pack_w<<<(18 * 2048 + 255) / 256, 256, 0, stream>>>(W2, wp2, 64);
  k_pack_w<<<( 9 * 2048 + 255) / 256, 256, 0, stream>>>(W3, wp3, 32);
  k_pack_w<<<(18 * 2048 + 255) / 256, 256, 0, stream>>>(W4, wp4, 64);

  const int tiles = NN / 16;          // 12500 row tiles
  const int wpb   = 8;                // 8 waves (256 threads) per block
  dim3 cgrid((tiles + wpb - 1) / wpb), cblk(wpb * 32);

  // Shortcut branch: conv1(1x3x3) -> lrelu -> BN -> conv2(3x1x3) -> lrelu -> BN
  k_conv<32><<<cgrid, cblk, 0, stream>>>(xbf, nbr133, wp1, yA, st1);
  k_bn_fin<<<1, 64, 0, stream>>>(st1, g0, b0);
  { int tot = (NN + 1) * 64;
    k_bn_apply<<<(tot + 255) / 256, 256, 0, stream>>>(yA, st1, act); }
  k_conv<64><<<cgrid, cblk, 0, stream>>>(act, nbr313, wp2, yB, st2);
  k_bn_fin<<<1, 64, 0, stream>>>(st2, g02, b02);

  // resA branch: conv3(3x1x3) -> lrelu -> BN -> conv4(1x3x3) -> lrelu -> BN
  k_conv<32><<<cgrid, cblk, 0, stream>>>(xbf, nbr313, wp3, yA, st3);
  k_bn_fin<<<1, 64, 0, stream>>>(st3, g1, b1);
  { int tot = (NN + 1) * 64;
    k_bn_apply<<<(tot + 255) / 256, 256, 0, stream>>>(yA, st3, act); }
  k_conv<64><<<cgrid, cblk, 0, stream>>>(act, nbr133, wp4, yA, st4);
  k_bn_fin<<<1, 64, 0, stream>>>(st4, g2, b2);

  // out = BN(conv4) + BN(conv2)
  { int tot = NN * 64;
    k_final<<<(tot + 255) / 256, 256, 0, stream>>>(yB, st2, yA, st4, out); }
}
